// CrossAttnVDTBlock_35708358099091
// MI455X (gfx1250) — compile-verified, matrix operand order checked by reading, not run on hardware
//
#include <hip/hip_runtime.h>
#include <hip/hip_bf16.h>
#include <math.h>

// ---------------- CDNA5 WMMA types ----------------
typedef __attribute__((ext_vector_type(16))) _Float16 v16h;
typedef __attribute__((ext_vector_type(8)))  _Float16 v8h;
typedef __attribute__((ext_vector_type(8)))  float    v8f;

union V16 { v16h v; v8h h[2]; };

// ---------------- problem constants ----------------
#define Dm    768
#define DFFm  3072
#define HEADS 12
#define HD    64
#define SEQ   2048      // T*N per batch
#define TOK   4096      // B*T*N total tokens
#define NF    8         // frames
#define ADA   4608      // 6*D

// Flip to 0 if the assembler rejects the async mnemonics.
#define USE_ASYNC_STAGE 1

__device__ __forceinline__ float gelu_tanh(float x) {
    float x3 = x * x * x;
    return 0.5f * x * (1.0f + tanhf(0.7978845608028654f * (x + 0.044715f * x3)));
}

// 16-byte global -> LDS stage (CDNA5 async data path; ASYNCcnt-tracked)
__device__ __forceinline__ void stage16(const _Float16* g, _Float16* l) {
#if USE_ASYNC_STAGE
    unsigned lds_off = (unsigned)(uintptr_t)l;   // LDS aperture: addr[31:0] == LDS byte offset
    asm volatile("global_load_async_to_lds_b128 %0, %1, off"
                 :: "v"(lds_off), "v"(g) : "memory");
#else
    *(v8h*)l = *(const v8h*)g;
#endif
}
__device__ __forceinline__ void wait_stage() {
#if USE_ASYNC_STAGE
    asm volatile("s_wait_asynccnt 0x0" ::: "memory");
#endif
}

// row-of-16 butterfly max on the VALU (v_permlane16_b32), no LDS traffic
__device__ __forceinline__ float permlane_step(float x, unsigned lo, unsigned hi) {
#if __has_builtin(__builtin_amdgcn_permlane16)
    int xi = __builtin_bit_cast(int, x);
    int r  = __builtin_amdgcn_permlane16(xi, xi, lo, hi, false, false);
    return __builtin_bit_cast(float, r);
#else
    (void)lo; (void)hi;
    return x;
#endif
}
__device__ __forceinline__ float rowmax16(float x) {
#if __has_builtin(__builtin_amdgcn_permlane16)
    x = fmaxf(x, permlane_step(x, 0x67452301u, 0xEFCDAB89u)); // xor 1
    x = fmaxf(x, permlane_step(x, 0x54761032u, 0xDCFE98BAu)); // xor 2
    x = fmaxf(x, permlane_step(x, 0x32107654u, 0xBA98FEDCu)); // xor 4
    x = fmaxf(x, permlane_step(x, 0xFEDCBA98u, 0x76543210u)); // xor 8
#else
    x = fmaxf(x, __shfl_xor(x, 1));
    x = fmaxf(x, __shfl_xor(x, 2));
    x = fmaxf(x, __shfl_xor(x, 4));
    x = fmaxf(x, __shfl_xor(x, 8));
#endif
    return x;
}

// =====================================================================
// WMMA GEMM:  C[M,N] = A[M,K](f16) * BwT[N,K](f16, pre-transposed) + bias
// EPI 0: out16 = f16(acc+bias)
// EPI 1: out16 = f16(gelu_tanh(acc+bias))
// EPI 2: outf += gate[batch,col] * (acc+bias)   (gate==nullptr -> 1.0)
// Tile: 128x128, BK=32, 8 waves, each wave 32x64 (2x4 WMMA frags)
// =====================================================================
template<int EPI>
__global__ __launch_bounds__(256) void gemm_wmma(
    const _Float16* __restrict__ A, const _Float16* __restrict__ BwT,
    const float* __restrict__ bias, const float* __restrict__ gate,
    _Float16* __restrict__ out16, float* __restrict__ outf,
    int M, int Nn, int K)
{
    __shared__ _Float16 sA[128 * 32];   // [row][k]
    __shared__ _Float16 sB[128 * 32];   // [n][k]

    const int t    = threadIdx.x;
    const int lane = t & 31, g = lane >> 4, l16 = lane & 15;
    const int wave = t >> 5, wm = wave & 3, wn = wave >> 2;
    const int rowT = blockIdx.y * 128;
    const int colT = blockIdx.x * 128;

    v8f acc[2][4];
    #pragma unroll
    for (int i = 0; i < 2; i++)
        #pragma unroll
        for (int j = 0; j < 4; j++)
            #pragma unroll
            for (int e = 0; e < 8; e++) acc[i][j][e] = 0.0f;

    const int arow = t >> 1;          // 0..127
    const int akh  = (t & 1) << 4;    // 0 or 16 halves

    for (int k0 = 0; k0 < K; k0 += 32) {
        const _Float16* asrc = A   + (size_t)(rowT + arow) * K + k0 + akh;
        const _Float16* bsrc = BwT + (size_t)(colT + arow) * K + k0 + akh;
        stage16(asrc,     &sA[arow * 32 + akh]);
        stage16(asrc + 8, &sA[arow * 32 + akh + 8]);
        stage16(bsrc,     &sB[arow * 32 + akh]);
        stage16(bsrc + 8, &sB[arow * 32 + akh + 8]);
        if (k0 + 32 < K) {
            __builtin_prefetch(asrc + 32, 0, 1);
            __builtin_prefetch(bsrc + 32, 0, 1);
        }
        wait_stage();
        __syncthreads();

        V16 af[2], bf[4];
        #pragma unroll
        for (int ms = 0; ms < 2; ms++) {
            int base = (wm * 32 + ms * 16 + l16) * 32;
            af[ms].h[0] = *(const v8h*)&sA[base + g * 8];
            af[ms].h[1] = *(const v8h*)&sA[base + 16 + g * 8];
        }
        #pragma unroll
        for (int ns = 0; ns < 4; ns++) {
            int base = (wn * 64 + ns * 16 + l16) * 32;
            bf[ns].h[0] = *(const v8h*)&sB[base + g * 8];
            bf[ns].h[1] = *(const v8h*)&sB[base + 16 + g * 8];
        }
        #pragma unroll
        for (int ms = 0; ms < 2; ms++)
            #pragma unroll
            for (int ns = 0; ns < 4; ns++)
                acc[ms][ns] = __builtin_amdgcn_wmma_f32_16x16x32_f16(
                    false, af[ms].v, false, bf[ns].v, (short)0, acc[ms][ns], false, false);
        __syncthreads();
    }

    // epilogue
    #pragma unroll
    for (int ms = 0; ms < 2; ms++)
        #pragma unroll
        for (int ns = 0; ns < 4; ns++)
            #pragma unroll
            for (int i = 0; i < 8; i++) {
                int r = rowT + wm * 32 + ms * 16 + g * 8 + i;
                int c = colT + wn * 64 + ns * 16 + l16;
                float v = acc[ms][ns][i] + bias[c];
                if (EPI == 1) v = gelu_tanh(v);
                if (EPI <= 1) {
                    out16[(size_t)r * Nn + c] = (_Float16)v;
                } else {
                    float gt = gate ? gate[(r >> 11) * ADA + c] : 1.0f;
                    outf[(size_t)r * Nn + c] += gt * v;
                }
            }
}

// =====================================================================
// WMMA flash self-attention (frame-causal).  One wave per 16-query block.
// Q,K f16 (token, h*64+d);  V pre-transposed vt[(b*H+h)*64+d][key] f16.
// Row-sum of P is fused into the matrix pipe via a P x ones WMMA, so the
// softmax denominator lands in the same C-layout rows as O.
// =====================================================================
__global__ __launch_bounds__(128) void flash_attn_kernel(
    const _Float16* __restrict__ q, const _Float16* __restrict__ k,
    const _Float16* __restrict__ vt, _Float16* __restrict__ o)
{
    __shared__ _Float16 pshare[4][16 * 32];   // per-wave P staging

    const int t    = threadIdx.x;
    const int wave = t >> 5;
    const int lane = t & 31, g = lane >> 4, l16 = lane & 15;
    const int h = blockIdx.y, b = blockIdx.z;
    const int qblk  = blockIdx.x * 4 + wave;
    const int qrow0 = qblk * 16;
    const int qf    = qrow0 >> 8;             // frame of this query block
    const int nkeys = (qf + 1) << 8;          // visible keys (multiple of 32)

    // Q fragments, pre-scaled by hd^-0.5 = 0.125
    const _Float16* qb = q + ((size_t)(b * SEQ + qrow0 + l16)) * Dm + h * HD;
    V16 aq[2];
    #pragma unroll
    for (int ks = 0; ks < 2; ks++) {
        aq[ks].h[0] = *(const v8h*)(qb + ks * 32 + g * 8);
        aq[ks].h[1] = *(const v8h*)(qb + ks * 32 + 16 + g * 8);
        #pragma unroll
        for (int e = 0; e < 16; e++) aq[ks].v[e] = aq[ks].v[e] * (_Float16)0.125f;
    }
    V16 ones;
    #pragma unroll
    for (int e = 0; e < 16; e++) ones.v[e] = (_Float16)1.0f;

    float m[8];
    v8f oacc[4], sumacc;
    #pragma unroll
    for (int i = 0; i < 8; i++) { m[i] = -1e30f; sumacc[i] = 0.0f; }
    #pragma unroll
    for (int ns = 0; ns < 4; ns++)
        #pragma unroll
        for (int i = 0; i < 8; i++) oacc[ns][i] = 0.0f;

    const _Float16* kb = k + (size_t)(b * SEQ) * Dm + h * HD;
    const _Float16* vb = vt + ((size_t)(b * HEADS + h) * HD) * SEQ;
    _Float16* ps = pshare[wave];

    for (int key0 = 0; key0 < nkeys; key0 += 32) {
        // ---- S = (Q*scale) K^T : 16x32 scores in two 16x16 f32 frags ----
        v8f s[2];
        #pragma unroll
        for (int ns = 0; ns < 2; ns++) {
            #pragma unroll
            for (int i = 0; i < 8; i++) s[ns][i] = 0.0f;
            #pragma unroll
            for (int ks = 0; ks < 2; ks++) {
                V16 kf;
                const _Float16* kp = kb + (size_t)(key0 + ns * 16 + l16) * Dm + ks * 32;
                kf.h[0] = *(const v8h*)(kp + g * 8);
                kf.h[1] = *(const v8h*)(kp + 16 + g * 8);
                s[ns] = __builtin_amdgcn_wmma_f32_16x16x32_f16(
                    false, aq[ks].v, false, kf.v, (short)0, s[ns], false, false);
            }
        }
        // ---- online softmax: row max via VALU permlane16 butterflies ----
        float mn[8], alpha[8];
        #pragma unroll
        for (int i = 0; i < 8; i++) {
            float mv = rowmax16(fmaxf(s[0][i], s[1][i]));
            mn[i]    = fmaxf(m[i], mv);
            alpha[i] = __expf(m[i] - mn[i]);
            m[i]     = mn[i];
            s[0][i]  = __expf(s[0][i] - mn[i]);
            s[1][i]  = __expf(s[1][i] - mn[i]);
        }
        #pragma unroll
        for (int ns = 0; ns < 4; ns++)
            #pragma unroll
            for (int i = 0; i < 8; i++) oacc[ns][i] *= alpha[i];
        #pragma unroll
        for (int i = 0; i < 8; i++) sumacc[i] *= alpha[i];

        // ---- P: C-layout -> A-layout via per-wave LDS round trip ----
        #pragma unroll
        for (int ns = 0; ns < 2; ns++)
            #pragma unroll
            for (int i = 0; i < 8; i++)
                ps[(g * 8 + i) * 32 + ns * 16 + l16] = (_Float16)s[ns][i];
        asm volatile("s_wait_dscnt 0x0" ::: "memory");  // wave-local RAW fence
        V16 pa;
        pa.h[0] = *(const v8h*)&ps[l16 * 32 + g * 8];
        pa.h[1] = *(const v8h*)&ps[l16 * 32 + 16 + g * 8];

        // ---- O += P V ;  denom += P * ones ----
        #pragma unroll
        for (int ns = 0; ns < 4; ns++) {
            V16 vf;
            const _Float16* vp = vb + (size_t)(ns * 16 + l16) * SEQ + key0;
            vf.h[0] = *(const v8h*)(vp + g * 8);
            vf.h[1] = *(const v8h*)(vp + 16 + g * 8);
            oacc[ns] = __builtin_amdgcn_wmma_f32_16x16x32_f16(
                false, pa.v, false, vf.v, (short)0, oacc[ns], false, false);
        }
        sumacc = __builtin_amdgcn_wmma_f32_16x16x32_f16(
            false, pa.v, false, ones.v, (short)0, sumacc, false, false);
        asm volatile("s_wait_dscnt 0x0" ::: "memory");  // WAR before next chunk
    }

    float inv[8];
    #pragma unroll
    for (int i = 0; i < 8; i++) inv[i] = 1.0f / sumacc[i];
    #pragma unroll
    for (int ns = 0; ns < 4; ns++)
        #pragma unroll
        for (int i = 0; i < 8; i++) {
            int r = qrow0 + g * 8 + i;
            int c = h * HD + ns * 16 + l16;
            o[((size_t)(b * SEQ + r)) * Dm + c] = (_Float16)(oacc[ns][i] * inv[i]);
        }
}

// ===================== small helper kernels ==========================

__global__ __launch_bounds__(256) void copy_kernel(
    const float* __restrict__ a, float* __restrict__ b, int n)
{
    int i = blockIdx.x * 256 + threadIdx.x;
    if (i < n) b[i] = a[i];
}

// f32 [K][N] weights -> f16 [N][K] (transposed, WMMA-ready)
__global__ __launch_bounds__(256) void wtrans_kernel(
    const float* __restrict__ w, _Float16* __restrict__ wT, int K, int Nn)
{
    int gid = blockIdx.x * 256 + threadIdx.x;
    if (gid >= K * Nn) return;
    int n = gid % Nn, kk = gid / Nn;
    wT[(size_t)n * K + kk] = (_Float16)w[gid];
}

// mods = silu(t) @ w_ada + b_ada      (2 x 4608)
__global__ __launch_bounds__(256) void mods_kernel(
    const float* __restrict__ tvec, const float* __restrict__ w_ada,
    const float* __restrict__ b_ada, float* __restrict__ mods)
{
    __shared__ float st[Dm];
    int r = blockIdx.y, tid = threadIdx.x;
    for (int i = tid; i < Dm; i += 256) {
        float v = tvec[r * Dm + i];
        st[i] = v / (1.0f + __expf(-v));
    }
    __syncthreads();
    int col = blockIdx.x * 256 + tid;
    float acc = b_ada[col];
    for (int kk = 0; kk < Dm; kk++) acc += st[kk] * w_ada[(size_t)kk * ADA + col];
    mods[r * ADA + col] = acc;
}

// LayerNorm (+ optional adaLN modulation) -> f16
__global__ __launch_bounds__(256) void ln_mod_kernel(
    const float* __restrict__ x, const float* __restrict__ mods,
    int shOff, int scOff, int useMod, _Float16* __restrict__ out)
{
    __shared__ float r1[256], r2[256];
    int row = blockIdx.x, tid = threadIdx.x;
    const float* xr = x + (size_t)row * Dm;
    float v0 = xr[tid], v1 = xr[tid + 256], v2 = xr[tid + 512];
    r1[tid] = v0 + v1 + v2;
    r2[tid] = v0 * v0 + v1 * v1 + v2 * v2;
    __syncthreads();
    for (int o = 128; o > 0; o >>= 1) {
        if (tid < o) { r1[tid] += r1[tid + o]; r2[tid] += r2[tid + o]; }
        __syncthreads();
    }
    float mu  = r1[0] * (1.0f / Dm);
    float var = r2[0] * (1.0f / Dm) - mu * mu;
    float inv = rsqrtf(var + 1e-6f);
    int b = row >> 11;
    float vs[3] = { v0, v1, v2 };
    #pragma unroll
    for (int j = 0; j < 3; j++) {
        int c = tid + j * 256;
        float y = (vs[j] - mu) * inv;
        if (useMod)
            y = y * (1.0f + mods[b * ADA + scOff + c]) + mods[b * ADA + shOff + c];
        out[(size_t)row * Dm + c] = (_Float16)y;
    }
}

// K/V projections of conditioning z (16 rows only)
__global__ __launch_bounds__(256) void cross_kv_kernel(
    const float* __restrict__ z,
    const float* __restrict__ wk, const float* __restrict__ bk,
    const float* __restrict__ wv, const float* __restrict__ bv,
    float* __restrict__ kz, float* __restrict__ vz)
{
    __shared__ float zr[Dm];
    int row = blockIdx.y, tid = threadIdx.x;
    for (int i = tid; i < Dm; i += 256) zr[i] = z[(size_t)row * Dm + i];
    __syncthreads();
    int col = blockIdx.x * 256 + tid;
    const float* w  = blockIdx.z ? wv : wk;
    const float* bb = blockIdx.z ? bv : bk;
    float acc = bb[col];
    for (int kk = 0; kk < Dm; kk++) acc += zr[kk] * w[(size_t)kk * Dm + col];
    float* dst = blockIdx.z ? vz : kz;
    dst[(size_t)row * Dm + col] = acc;
}

// cross attention: <=8 keys per query, frame-causal over frames
__global__ __launch_bounds__(256) void cross_attn_kernel(
    const _Float16* __restrict__ q, const float* __restrict__ kz,
    const float* __restrict__ vz, _Float16* __restrict__ o)
{
    int gid   = blockIdx.x * 256 + threadIdx.x;   // TOK*HEADS threads
    int token = gid / HEADS;
    int h     = gid % HEADS;
    int b     = token >> 11;
    int qf    = (token >> 8) & 7;
    const _Float16* qp = q + (size_t)token * Dm + h * HD;
    float s[NF];
    float mx = -1e30f;
    for (int kt = 0; kt <= qf; kt++) {
        const float* kp = kz + (size_t)(b * NF + kt) * Dm + h * HD;
        float acc = 0.0f;
        for (int d2 = 0; d2 < HD; d2++) acc += (float)qp[d2] * kp[d2];
        s[kt] = acc * 0.125f;
        mx = fmaxf(mx, s[kt]);
    }
    float den = 0.0f;
    for (int kt = 0; kt <= qf; kt++) { s[kt] = __expf(s[kt] - mx); den += s[kt]; }
    float invd = 1.0f / den;
    for (int d2 = 0; d2 < HD; d2++) {
        float acc = 0.0f;
        for (int kt = 0; kt <= qf; kt++)
            acc += s[kt] * vz[(size_t)(b * NF + kt) * Dm + h * HD + d2];
        o[(size_t)token * Dm + h * HD + d2] = (_Float16)(acc * invd);
    }
}

// transpose V (token, h*64+d) -> vt[(b*H+h)*64+d][key]
__global__ __launch_bounds__(256) void vtrans_kernel(
    const _Float16* __restrict__ v, _Float16* __restrict__ vt)
{
    int gid = blockIdx.x * 256 + threadIdx.x;   // 2*12*2048*64 = 3145728
    int d2  = gid & 63;
    int key = (gid >> 6) & 2047;
    int bh  = gid >> 17;                        // 0..23
    int b   = bh / HEADS, h = bh % HEADS;
    vt[((size_t)bh * HD + d2) * SEQ + key] =
        v[((size_t)(b * SEQ) + key) * Dm + h * HD + d2];
}

// ========================= host launch ===============================
extern "C" void kernel_launch(void* const* d_in, const int* in_sizes, int n_in,
                              void* d_out, int out_size, void* d_ws, size_t ws_size,
                              hipStream_t stream)
{
    const float* x     = (const float*)d_in[0];
    const float* z     = (const float*)d_in[1];
    const float* tv    = (const float*)d_in[2];
    const float* w_ada = (const float*)d_in[3];
    const float* b_ada = (const float*)d_in[4];
    const float* c_wq  = (const float*)d_in[5];  const float* c_bq = (const float*)d_in[6];
    const float* c_wk  = (const float*)d_in[7];  const float* c_bk = (const float*)d_in[8];
    const float* c_wv  = (const float*)d_in[9];  const float* c_bv = (const float*)d_in[10];
    const float* c_wo  = (const float*)d_in[11]; const float* c_bo = (const float*)d_in[12];
    const float* w_fc1 = (const float*)d_in[13]; const float* b_fc1 = (const float*)d_in[14];
    const float* s_wq  = (const float*)d_in[15]; const float* s_bq = (const float*)d_in[16];
    const float* s_wk  = (const float*)d_in[17]; const float* s_bk = (const float*)d_in[18];
    const float* s_wv  = (const float*)d_in[19]; const float* s_bv = (const float*)d_in[20];
    const float* s_wo  = (const float*)d_in[21]; const float* s_bo = (const float*)d_in[22];
    const float* w_fc2 = (const float*)d_in[23]; const float* b_fc2 = (const float*)d_in[24];
    const float* m_w1  = (const float*)d_in[25]; const float* m_b1 = (const float*)d_in[26];
    const float* m_w2  = (const float*)d_in[27]; const float* m_b2 = (const float*)d_in[28];
    (void)in_sizes; (void)n_in; (void)out_size; (void)ws_size;

    float* out = (float*)d_out;

    // carve workspace (256B aligned)
    char* ws = (char*)d_ws;
    size_t off = 0;
    auto carve = [&](size_t bytes) -> char* {
        char* p = ws + off;
        off += (bytes + 255) & ~(size_t)255;
        return p;
    };
    float*     mods  = (float*)    carve((size_t)2 * ADA * 4);
    _Float16*  buf0  = (_Float16*) carve((size_t)TOK * Dm * 2);   // LN/activation f16
    _Float16*  buf1  = (_Float16*) carve((size_t)TOK * Dm * 2);   // Q
    _Float16*  buf2  = (_Float16*) carve((size_t)TOK * Dm * 2);   // K
    _Float16*  buf3  = (_Float16*) carve((size_t)TOK * Dm * 2);   // V
    _Float16*  buf4  = (_Float16*) carve((size_t)TOK * Dm * 2);   // attn out
    _Float16*  bufvt = (_Float16*) carve((size_t)TOK * Dm * 2);   // V transposed
    _Float16*  buf5  = (_Float16*) carve((size_t)TOK * DFFm * 2); // MLP hidden
    float*     kz    = (float*)    carve((size_t)16 * Dm * 4);
    float*     vz    = (float*)    carve((size_t)16 * Dm * 4);
    // pre-transposed f16 weights [N][K]
    _Float16*  wt_cq  = (_Float16*) carve((size_t)Dm * Dm * 2);
    _Float16*  wt_co  = (_Float16*) carve((size_t)Dm * Dm * 2);
    _Float16*  wt_fc1 = (_Float16*) carve((size_t)Dm * Dm * 2);
    _Float16*  wt_sq  = (_Float16*) carve((size_t)Dm * Dm * 2);
    _Float16*  wt_sk  = (_Float16*) carve((size_t)Dm * Dm * 2);
    _Float16*  wt_sv  = (_Float16*) carve((size_t)Dm * Dm * 2);
    _Float16*  wt_so  = (_Float16*) carve((size_t)Dm * Dm * 2);
    _Float16*  wt_fc2 = (_Float16*) carve((size_t)Dm * Dm * 2);
    _Float16*  wt_m1  = (_Float16*) carve((size_t)Dm * DFFm * 2);
    _Float16*  wt_m2  = (_Float16*) carve((size_t)DFFm * Dm * 2);

    const int GRID_DD = (Dm * Dm) / 256;      // 2304
    const int GRID_DF = (Dm * DFFm) / 256;    // 9216
    wtrans_kernel<<<GRID_DD, 256, 0, stream>>>(c_wq,  wt_cq,  Dm, Dm);
    wtrans_kernel<<<GRID_DD, 256, 0, stream>>>(c_wo,  wt_co,  Dm, Dm);
    wtrans_kernel<<<GRID_DD, 256, 0, stream>>>(w_fc1, wt_fc1, Dm, Dm);
    wtrans_kernel<<<GRID_DD, 256, 0, stream>>>(s_wq,  wt_sq,  Dm, Dm);
    wtrans_kernel<<<GRID_DD, 256, 0, stream>>>(s_wk,  wt_sk,  Dm, Dm);
    wtrans_kernel<<<GRID_DD, 256, 0, stream>>>(s_wv,  wt_sv,  Dm, Dm);
    wtrans_kernel<<<GRID_DD, 256, 0, stream>>>(s_wo,  wt_so,  Dm, Dm);
    wtrans_kernel<<<GRID_DD, 256, 0, stream>>>(w_fc2, wt_fc2, Dm, Dm);
    wtrans_kernel<<<GRID_DF, 256, 0, stream>>>(m_w1,  wt_m1,  Dm, DFFm);
    wtrans_kernel<<<GRID_DF, 256, 0, stream>>>(m_w2,  wt_m2,  DFFm, Dm);

    auto gemmF16 = [&](const _Float16* A, const _Float16* Bt, const float* bias,
                       _Float16* O, int M, int Nn, int K) {
        gemm_wmma<0><<<dim3(Nn / 128, M / 128), 256, 0, stream>>>(
            A, Bt, bias, nullptr, O, nullptr, M, Nn, K);
    };
    auto gemmGelu = [&](const _Float16* A, const _Float16* Bt, const float* bias,
                        _Float16* O, int M, int Nn, int K) {
        gemm_wmma<1><<<dim3(Nn / 128, M / 128), 256, 0, stream>>>(
            A, Bt, bias, nullptr, O, nullptr, M, Nn, K);
    };
    auto gemmAcc = [&](const _Float16* A, const _Float16* Bt, const float* bias,
                       const float* gate, float* O, int M, int Nn, int K) {
        gemm_wmma<2><<<dim3(Nn / 128, M / 128), 256, 0, stream>>>(
            A, Bt, bias, gate, nullptr, O, M, Nn, K);
    };

    // chunk offsets in mods rows: sh_s, sc_s, g_s, sh_m, sc_m, g_m
    const int SH_S = 0, SC_S = 768, G_S = 1536, SH_M = 2304, SC_M = 3072, G_M = 3840;

    // running x lives in d_out (fp32 residual stream)
    copy_kernel<<<(TOK * Dm) / 256, 256, 0, stream>>>(x, out, TOK * Dm);
    mods_kernel<<<dim3(ADA / 256, 2), 256, 0, stream>>>(tv, w_ada, b_ada, mods);

    // ---- cross attention + fc1 residual ----
    ln_mod_kernel<<<TOK, 256, 0, stream>>>(out, mods, 0, 0, 0, buf0);
    gemmF16(buf0, wt_cq, c_bq, buf1, TOK, Dm, Dm);
    cross_kv_kernel<<<dim3(Dm / 256, 16, 2), 256, 0, stream>>>(
        z, c_wk, c_bk, c_wv, c_bv, kz, vz);
    cross_attn_kernel<<<(TOK * HEADS) / 256, 256, 0, stream>>>(buf1, kz, vz, buf4);
    gemmF16(buf4, wt_co, c_bo, buf0, TOK, Dm, Dm);
    gemmAcc(buf0, wt_fc1, b_fc1, nullptr, out, TOK, Dm, Dm);

    // ---- modulated frame-causal self attention + gated fc2 residual ----
    ln_mod_kernel<<<TOK, 256, 0, stream>>>(out, mods, SH_S, SC_S, 1, buf0);
    gemmF16(buf0, wt_sq, s_bq, buf1, TOK, Dm, Dm);
    gemmF16(buf0, wt_sk, s_bk, buf2, TOK, Dm, Dm);
    gemmF16(buf0, wt_sv, s_bv, buf3, TOK, Dm, Dm);
    vtrans_kernel<<<(TOK * Dm) / 256, 256, 0, stream>>>(buf3, bufvt);
    flash_attn_kernel<<<dim3(32, HEADS, 2), 128, 0, stream>>>(buf1, buf2, bufvt, buf4);
    gemmF16(buf4, wt_so, s_bo, buf0, TOK, Dm, Dm);
    gemmAcc(buf0, wt_fc2, b_fc2, mods + G_S, out, TOK, Dm, Dm);

    // ---- modulated MLP (tanh-GELU) + gated residual ----
    ln_mod_kernel<<<TOK, 256, 0, stream>>>(out, mods, SH_M, SC_M, 1, buf0);
    gemmGelu(buf0, wt_m1, m_b1, buf5, TOK, DFFm, Dm);
    gemmAcc(buf5, wt_m2, m_b2, mods + G_M, out, TOK, Dm, DFFm);
}
